// VS_DiT_29678224015666
// MI455X (gfx1250) — matrix operand order repaired and checked by reading, not compiled, not run on hardware
//
#include <hip/hip_runtime.h>
#include <cstdint>

#define HIDN 1024
#define NHEADS 16
#define DKH 64
#define LATD 512
#define CTXD 768
#define NBLK 4
#define MLPH 4096
#define BB 4
#define LL 1024
#define LCC 256

typedef __bf16 bf16_t;
typedef bf16_t v16bf __attribute__((ext_vector_type(16)));
typedef float v8f __attribute__((ext_vector_type(8)));
typedef unsigned short u16;
typedef int v4i_vs __attribute__((vector_size(16)));  // matches builtin param

union Frag { v16bf v; uint4 q[2]; };

#if defined(__AMDGCN__) && __has_builtin(__builtin_amdgcn_global_load_async_to_lds_b128) && \
    __has_builtin(__builtin_amdgcn_s_wait_asynccnt)
#define HAVE_ASYNC_LDS 1
#else
#define HAVE_ASYNC_LDS 0
#endif

__device__ inline void async_ld_b128(u16* lds_dst, const u16* gsrc) {
#if HAVE_ASYNC_LDS
  __builtin_amdgcn_global_load_async_to_lds_b128(
      (__attribute__((address_space(1))) v4i_vs*)(uintptr_t)gsrc,
      (__attribute__((address_space(3))) v4i_vs*)(uintptr_t)lds_dst, 0, 0);
#else
  *(uint4*)lds_dst = *(const uint4*)gsrc;
#endif
}
__device__ inline void wait_async_all() {
#if HAVE_ASYNC_LDS
  __builtin_amdgcn_s_wait_asynccnt(0);
#endif
}

__device__ inline u16 f2bf(float f) {
  unsigned u = __float_as_uint(f);
  unsigned r = (u + 0x7FFFu + ((u >> 16) & 1u)) >> 16;
  return (u16)r;
}
__device__ inline v8f vzero() {
  v8f z;
#pragma unroll
  for (int i = 0; i < 8; i++) z[i] = 0.f;
  return z;
}
__device__ inline v8f wmma_bf16(v16bf a, v16bf b, v8f c) {
  return __builtin_amdgcn_wmma_f32_16x16x32_bf16(false, a, false, b, (short)0, c,
                                                 false, false);
}
__device__ inline float gelu_tanh(float v) {
  const float kk = 0.7978845608028654f;
  return 0.5f * v * (1.f + tanhf(kk * (v + 0.044715f * v * v * v)));
}
__device__ inline float silu_f(float v) { return v / (1.f + __expf(-v)); }

// ---------------------------------------------------------------------------
// GEMM: C[M,N] = epilogue( A_bf16[M,K] * W_f32[K,N] )
// 128x128 tile, BK=32, 8 waves (4 M-waves x 2 N-waves), wmma bf16.
// Double-buffered LDS; A tile staged via async global->LDS (ASYNCcnt),
// W tile converted fp32->bf16 and stored transposed (contiguous-K B frags).
// epilogue: v=(acc+bias); act; v*=outScale; v*=gate[b,n]; v+=resid; f32/bf16.
// Requires M % 128 == 0 (true for all call sites).
// ---------------------------------------------------------------------------
#define GBM 128
#define GBN 128
#define GBK 32
#define GLDA 40

__global__ __launch_bounds__(256) void k_gemm(
    const u16* __restrict__ A, const float* __restrict__ W,
    const float* __restrict__ bias, float* __restrict__ outF,
    u16* __restrict__ outB, const float* __restrict__ resid,
    const float* __restrict__ gate, int gStride, int rowsPerBatch, int M, int N,
    int K, int act, float outScale) {
  __shared__ u16 As[2][GBM * GLDA];
  __shared__ u16 Bs[2][GBN * GLDA];
  const int tid = threadIdx.x;
  const int wave = tid >> 5, lane = tid & 31;
  const int g = lane >> 4, lr = lane & 15;
  const int wM = wave & 3, wN = wave >> 2;
  const int m0 = blockIdx.y * GBM, n0 = blockIdx.x * GBN;

  v8f acc[2][4];
#pragma unroll
  for (int mt = 0; mt < 2; mt++)
#pragma unroll
    for (int nt = 0; nt < 4; nt++) acc[mt][nt] = vzero();

  const int rA = tid >> 1;       // A staging: row 0..127
  const int qA = (tid & 1) * 2;  // 2 of 4 16B chunks per thread (x2 iters)
  const int nW = tid & 127;      // W staging: column inside tile
  const int khW = tid >> 7;      // W staging: k-half (0/1)

  // loop-invariant staging bases (pointer induction; avoids per-step mul64)
  const u16* aRow = A + (size_t)(m0 + rA) * K;             // + k0 + qd*8
  const float* wCol = W + (size_t)khW * 16 * N + n0 + nW;  // + k0*N, step N

  auto stageA = [&](int k0, int buf) {
#pragma unroll
    for (int i = 0; i < 2; i++) {
      int qd = qA + i;
      async_ld_b128(&As[buf][rA * GLDA + qd * 8], aRow + k0 + qd * 8);
    }
  };
  auto stageW = [&](int k0, int buf) {
    const float* wp = wCol + (size_t)k0 * N;
    u16 us[16];
#pragma unroll
    for (int kk = 0; kk < 16; kk++) {
      us[kk] = f2bf(*wp);
      wp += N;
    }
    *(uint4*)&Bs[buf][nW * GLDA + khW * 16] = *(uint4*)&us[0];
    *(uint4*)&Bs[buf][nW * GLDA + khW * 16 + 8] = *(uint4*)&us[8];
    if (k0 + GBK < K)
      __builtin_prefetch(wp + (size_t)(GBK - 16) * N, 0, 1);  // next tile
  };

  stageA(0, 0);
  stageW(0, 0);
  int buf = 0;
  for (int k0 = 0; k0 < K; k0 += GBK) {
    wait_async_all();
    __syncthreads();
    if (k0 + GBK < K) {  // stage next tile into the other buffer
      stageA(k0 + GBK, buf ^ 1);
      stageW(k0 + GBK, buf ^ 1);
    }
    // --- compute current tile ---
    Frag af[2], bf[4];
#pragma unroll
    for (int mt = 0; mt < 2; mt++) {
      int rr = wM * 32 + mt * 16 + lr;
      af[mt].q[0] = *(const uint4*)&As[buf][rr * GLDA + 16 * g];
      af[mt].q[1] = *(const uint4*)&As[buf][rr * GLDA + 16 * g + 8];
    }
#pragma unroll
    for (int nt = 0; nt < 4; nt++) {
      int nn = wN * 64 + nt * 16 + lr;
      bf[nt].q[0] = *(const uint4*)&Bs[buf][nn * GLDA + 16 * g];
      bf[nt].q[1] = *(const uint4*)&Bs[buf][nn * GLDA + 16 * g + 8];
    }
#pragma unroll
    for (int mt = 0; mt < 2; mt++)
#pragma unroll
      for (int nt = 0; nt < 4; nt++)
        acc[mt][nt] = wmma_bf16(af[mt].v, bf[nt].v, acc[mt][nt]);
    buf ^= 1;
  }

  // --- epilogue ---
#pragma unroll
  for (int mt = 0; mt < 2; mt++) {
#pragma unroll
    for (int nt = 0; nt < 4; nt++) {
      int col = n0 + wN * 64 + nt * 16 + lr;
      float bv = bias ? bias[col] : 0.f;
      int row0 = m0 + wM * 32 + mt * 16 + 8 * g;
#pragma unroll
      for (int r = 0; r < 8; r++) {
        int row = row0 + r;
        float v = acc[mt][nt][r] + bv;
        if (act == 1) v = gelu_tanh(v);
        v *= outScale;
        if (gate) v *= gate[(size_t)(row / rowsPerBatch) * gStride + col];
        if (resid) v += resid[(size_t)row * N + col];
        if (outF) outF[(size_t)row * N + col] = v;
        if (outB) outB[(size_t)row * N + col] = f2bf(v);
      }
    }
  }
}

// ---------------------------------------------------------------------------
// Flash attention (bf16 in/out, f32 accum). One WG = (b, h, 64 q rows),
// 4 waves x 16 q rows. K staged async into LDS; V transposed via VGPRs.
// Scale 1/sqrt(dk) is pre-folded into Q by the Wq GEMM epilogue.
// ---------------------------------------------------------------------------
#define ATP 72

__global__ __launch_bounds__(128) void k_attn(
    const u16* __restrict__ Q, const u16* __restrict__ Kx,
    const u16* __restrict__ V, u16* __restrict__ O,
    const unsigned char* __restrict__ maskp, int Lq, int Lk) {
  __shared__ u16 Ks[64 * ATP];
  __shared__ u16 Vt[64 * ATP];
  __shared__ u16 Ps[4 * 16 * ATP];
  const int b = blockIdx.z, h = blockIdx.y, qt = blockIdx.x;
  const int tid = threadIdx.x, wave = tid >> 5, lane = tid & 31;
  const int g = lane >> 4, lr = lane & 15;
  const size_t qrowBase = (size_t)b * Lq + qt * 64 + wave * 16;

  // Q fragments (held across the whole key loop)
  Frag qa[2];
  {
    const u16* qp = Q + (qrowBase + lr) * HIDN + h * 64;
#pragma unroll
    for (int ks = 0; ks < 2; ks++) {
      qa[ks].q[0] = *(const uint4*)(qp + ks * 32 + 16 * g);
      qa[ks].q[1] = *(const uint4*)(qp + ks * 32 + 16 * g + 8);
    }
  }

  v8f o[4];
#pragma unroll
  for (int i = 0; i < 4; i++) o[i] = vzero();
  float mrun[8], lrun[8];
#pragma unroll
  for (int r = 0; r < 8; r++) { mrun[r] = -1e30f; lrun[r] = 0.f; }

  const int nkb = Lk / 64;
  for (int kb = 0; kb < nkb; kb++) {
    __syncthreads();
    // stage K (async, row-major) and V (transposed) for this key block
#pragma unroll
    for (int i = 0; i < 4; i++) {
      int c = tid + i * 128;  // 0..511
      int kr = c >> 3, dq = c & 7;
      size_t krow = (size_t)b * Lk + (size_t)kb * 64 + kr;
      async_ld_b128(&Ks[kr * ATP + dq * 8], Kx + krow * HIDN + h * 64 + dq * 8);
      uint4 vd = *(const uint4*)(V + krow * HIDN + h * 64 + dq * 8);
      u16 tmp[8];
      *(uint4*)tmp = vd;
#pragma unroll
      for (int j = 0; j < 8; j++) Vt[(dq * 8 + j) * ATP + kr] = tmp[j];
    }
    wait_async_all();
    __syncthreads();

    // S = Q * K^T  (16 q rows x 64 keys per wave)
    v8f st[4];
#pragma unroll
    for (int nt = 0; nt < 4; nt++) st[nt] = vzero();
#pragma unroll
    for (int nt = 0; nt < 4; nt++) {
#pragma unroll
      for (int ks = 0; ks < 2; ks++) {
        Frag bfr;
        bfr.q[0] = *(const uint4*)&Ks[(nt * 16 + lr) * ATP + ks * 32 + 16 * g];
        bfr.q[1] =
            *(const uint4*)&Ks[(nt * 16 + lr) * ATP + ks * 32 + 16 * g + 8];
        st[nt] = wmma_bf16(qa[ks].v, bfr.v, st[nt]);
      }
    }
    if (maskp) {
#pragma unroll
      for (int nt = 0; nt < 4; nt++) {
        int key = kb * 64 + nt * 16 + lr;
        if (maskp[(size_t)b * Lk + key]) {
#pragma unroll
          for (int r = 0; r < 8; r++) st[nt][r] = -1e30f;
        }
      }
    }

    // online softmax (rows live in 16-lane groups; g preserved by xor<16)
#pragma unroll
    for (int r = 0; r < 8; r++) {
      float rmax = st[0][r];
#pragma unroll
      for (int nt = 1; nt < 4; nt++) rmax = fmaxf(rmax, st[nt][r]);
      for (int off = 1; off < 16; off <<= 1)
        rmax = fmaxf(rmax, __shfl_xor(rmax, off, 32));
      float mnew = fmaxf(mrun[r], rmax);
      float alpha = __expf(mrun[r] - mnew);
      mrun[r] = mnew;
      float rsum = 0.f;
#pragma unroll
      for (int nt = 0; nt < 4; nt++) {
        float p = __expf(st[nt][r] - mnew);
        st[nt][r] = p;
        rsum += p;
      }
      for (int off = 1; off < 16; off <<= 1) rsum += __shfl_xor(rsum, off, 32);
      lrun[r] = lrun[r] * alpha + rsum;
#pragma unroll
      for (int nt = 0; nt < 4; nt++) o[nt][r] *= alpha;
    }

    // P -> LDS (re-layout C-frag -> A-frag), then O += P * V
    u16* pw = &Ps[wave * 16 * ATP];
#pragma unroll
    for (int nt = 0; nt < 4; nt++)
#pragma unroll
      for (int r = 0; r < 8; r++)
        pw[(r + 8 * g) * ATP + nt * 16 + lr] = f2bf(st[nt][r]);
    __syncthreads();
    Frag pa[2];
#pragma unroll
    for (int ks = 0; ks < 2; ks++) {
      pa[ks].q[0] = *(const uint4*)&pw[lr * ATP + ks * 32 + 16 * g];
      pa[ks].q[1] = *(const uint4*)&pw[lr * ATP + ks * 32 + 16 * g + 8];
    }
#pragma unroll
    for (int ntd = 0; ntd < 4; ntd++) {
#pragma unroll
      for (int ks = 0; ks < 2; ks++) {
        Frag vb;
        vb.q[0] = *(const uint4*)&Vt[(ntd * 16 + lr) * ATP + ks * 32 + 16 * g];
        vb.q[1] =
            *(const uint4*)&Vt[(ntd * 16 + lr) * ATP + ks * 32 + 16 * g + 8];
        o[ntd] = wmma_bf16(pa[ks].v, vb.v, o[ntd]);
      }
    }
  }

  // finalize: O / l, write bf16
  float inv[8];
#pragma unroll
  for (int r = 0; r < 8; r++) inv[r] = 1.f / lrun[r];
#pragma unroll
  for (int ntd = 0; ntd < 4; ntd++) {
#pragma unroll
    for (int r = 0; r < 8; r++) {
      size_t row = qrowBase + r + 8 * g;
      int col = h * 64 + ntd * 16 + lr;
      O[row * HIDN + col] = f2bf(o[ntd][r] * inv[r]);
    }
  }
}

// ---------------------------------------------------------------------------
// LayerNorm (eps 1e-6) with optional AdaLN modulation, bf16 output.
// ---------------------------------------------------------------------------
__global__ __launch_bounds__(256) void k_ln(const float* __restrict__ x,
                                            const float* __restrict__ gm,
                                            const float* __restrict__ bt,
                                            int gStride, int rowsPerBatch, int D,
                                            u16* __restrict__ out) {
  __shared__ float red[256];
  const int row = blockIdx.x, tid = threadIdx.x;
  const float* xr = x + (size_t)row * D;
  float s = 0.f;
  for (int i = tid; i < D; i += 256) s += xr[i];
  red[tid] = s;
  __syncthreads();
  for (int st = 128; st > 0; st >>= 1) {
    if (tid < st) red[tid] += red[tid + st];
    __syncthreads();
  }
  float mean = red[0] / (float)D;
  __syncthreads();
  float v = 0.f;
  for (int i = tid; i < D; i += 256) {
    float d = xr[i] - mean;
    v += d * d;
  }
  red[tid] = v;
  __syncthreads();
  for (int st = 128; st > 0; st >>= 1) {
    if (tid < st) red[tid] += red[tid + st];
    __syncthreads();
  }
  float rstd = rsqrtf(red[0] / (float)D + 1e-6f);
  int b = row / rowsPerBatch;
  for (int i = tid; i < D; i += 256) {
    float y = (xr[i] - mean) * rstd;
    if (gm) y = y * (1.f + gm[(size_t)b * gStride + i]) + bt[(size_t)b * gStride + i];
    out[(size_t)row * D + i] = f2bf(y);
  }
}

// ---------------------------------------------------------------------------
// Elementwise / small kernels
// ---------------------------------------------------------------------------
__global__ void k_prep_xin(const float* __restrict__ z, u16* __restrict__ out,
                           int n) {
  int idx = blockIdx.x * blockDim.x + threadIdx.x;
  if (idx >= n) return;
  int m = idx / LATD, d = idx % LATD;
  int pos = m % LL;
  int j = d >> 1;
  float ang = (float)pos * __expf((float)(2 * j) * (-9.210340371976184f / (float)LATD));
  float pe = (d & 1) ? __cosf(ang) : __sinf(ang);
  out[idx] = f2bf(z[idx] + pe);
}

__global__ void k_rope(const float* __restrict__ y, float* __restrict__ x,
                       int n) {
  int idx = blockIdx.x * blockDim.x + threadIdx.x;
  if (idx >= n) return;  // n = rows * 512
  int m = idx / 512, i = idx % 512;
  int pos = m % LL;
  float invf = __expf(-9.210340371976184f * (float)i / 512.f);
  float c = __cosf((float)pos * invf), s = __sinf((float)pos * invf);
  float x1 = y[(size_t)m * HIDN + i];
  float x2 = y[(size_t)m * HIDN + 512 + i];
  x[(size_t)m * HIDN + i] = x1 * c - x2 * s;
  x[(size_t)m * HIDN + 512 + i] = x1 * s + x2 * c;
}

__global__ void k_timefreq(const float* __restrict__ t, float* __restrict__ out) {
  int idx = blockIdx.x * blockDim.x + threadIdx.x;
  if (idx >= BB * 256) return;
  int b = idx / 256, j2 = idx % 256;
  int j = (j2 < 128) ? j2 : (j2 - 128);
  float fr = __expf(-9.210340371976184f * (float)j / 128.f);
  float ta = t[b] * fr;
  out[idx] = (j2 < 128) ? __cosf(ta) : __sinf(ta);
}

__global__ void k_silu(const float* __restrict__ in, float* __restrict__ out,
                       int n) {
  int idx = blockIdx.x * blockDim.x + threadIdx.x;
  if (idx < n) out[idx] = silu_f(in[idx]);
}

__global__ void k_cast_bf16(const float* __restrict__ in, u16* __restrict__ out,
                            int n) {
  int idx = blockIdx.x * blockDim.x + threadIdx.x;
  if (idx < n) out[idx] = f2bf(in[idx]);
}

// tiny-M linear (f32 in/out): out[m,n] = act(sum_k in[m,k]*w[k,n] + b[n])
__global__ void k_small_linear(const float* __restrict__ in,
                               const float* __restrict__ w,
                               const float* __restrict__ b,
                               float* __restrict__ out, int K, int N, int act) {
  int n = blockIdx.x * 64 + threadIdx.x;
  int m = blockIdx.y;
  if (n >= N) return;
  float s = b[n];
  const float* ir = in + (size_t)m * K;
  const float* wp = w + n;
  for (int k = 0; k < K; k++) {
    s += ir[k] * (*wp);
    wp += N;
  }
  if (act == 1) s = silu_f(s);
  out[(size_t)m * N + n] = s;
}

// ---------------------------------------------------------------------------
// Host driver
// ---------------------------------------------------------------------------
extern "C" void kernel_launch(void* const* d_in, const int* in_sizes, int n_in,
                              void* d_out, int out_size, void* d_ws,
                              size_t ws_size, hipStream_t stream) {
  (void)in_sizes;
  (void)out_size;
  if (n_in < 112) return;

  const float* z_t = (const float*)d_in[0];
  const float* t_in = (const float*)d_in[1];
  const float* ctx_seq = (const float*)d_in[2];
  const unsigned char* pad = (const unsigned char*)d_in[3];
  auto PF = [&](int i) { return (const float*)d_in[i]; };

  // workspace layout
  char* wsB = (char*)d_ws;
  size_t off = 0;
  auto alloc = [&](size_t bytes) -> char* {
    char* p = wsB + off;
    off += (bytes + 255) & ~(size_t)255;
    return p;
  };
  const size_t TOK = (size_t)BB * LL;  // 4096
  u16* xin = (u16*)alloc(TOK * LATD * 2);
  u16* ctxA = (u16*)alloc((size_t)BB * LCC * CTXD * 2);
  u16* ctxB = (u16*)alloc((size_t)BB * LCC * HIDN * 2);
  float* ybuf = (float*)alloc(TOK * HIDN * 4);
  float* x = (float*)alloc(TOK * HIDN * 4);
  u16* hb = (u16*)alloc(TOK * HIDN * 2);
  u16* qb = (u16*)alloc(TOK * HIDN * 2);
  u16* kb = (u16*)alloc(TOK * HIDN * 2);
  u16* vb = (u16*)alloc(TOK * HIDN * 2);
  u16* attnb = (u16*)alloc(TOK * HIDN * 2);
  u16* m1 = (u16*)alloc(TOK * MLPH * 2);
  u16* m2 = (u16*)alloc(TOK * MLPH * 2);
  float* temb0 = (float*)alloc((size_t)BB * 256 * 4);
  float* h1 = (float*)alloc((size_t)BB * HIDN * 4);
  float* temb = (float*)alloc((size_t)BB * HIDN * 4);
  float* stb = (float*)alloc((size_t)BB * HIDN * 4);
  float* modb = (float*)alloc((size_t)BB * 6 * HIDN * 4);
  float* fmod = (float*)alloc((size_t)BB * 2 * HIDN * 4);
  if (off > ws_size) return;

  auto gemm = [&](const u16* A, const float* W, const float* bias, float* outF,
                  u16* outB, const float* resid, const float* gate, int gStride,
                  int M, int N, int K, int act, float outScale) {
    dim3 g(N / GBN, M / GBM);
    k_gemm<<<g, dim3(256), 0, stream>>>(A, W, bias, outF, outB, resid, gate,
                                        gStride, LL, M, N, K, act, outScale);
  };

  // --- input path ---
  {
    int n = (int)(TOK * LATD);
    k_prep_xin<<<(n + 255) / 256, 256, 0, stream>>>(z_t, xin, n);
  }
  {
    int n = (int)((size_t)BB * LCC * CTXD);
    k_cast_bf16<<<(n + 255) / 256, 256, 0, stream>>>(ctx_seq, ctxA, n);
  }
  // proj_in : [4096,512] x [512,1024]
  gemm(xin, PF(107), PF(106), ybuf, nullptr, nullptr, nullptr, 0, (int)TOK,
       HIDN, LATD, 0, 1.f);
  {
    int n = (int)(TOK * 512);
    k_rope<<<(n + 255) / 256, 256, 0, stream>>>(ybuf, x, n);
  }
  // t embedding chain (tiny)
  k_timefreq<<<4, 256, 0, stream>>>(t_in, temb0);
  k_small_linear<<<dim3(HIDN / 64, BB), 64, 0, stream>>>(temb0, PF(109),
                                                         PF(108), h1, 256,
                                                         HIDN, 1);
  k_small_linear<<<dim3(HIDN / 64, BB), 64, 0, stream>>>(h1, PF(111), PF(110),
                                                         temb, HIDN, HIDN, 0);
  k_silu<<<(BB * HIDN + 255) / 256, 256, 0, stream>>>(temb, stb, BB * HIDN);
  // ctx_proj : [1024,768] x [768,1024] -> bf16
  gemm(ctxA, PF(101), PF(100), nullptr, ctxB, nullptr, nullptr, 0, BB * LCC,
       HIDN, CTXD, 0, 1.f);

  const float scaleQ = 0.125f;  // 1/sqrt(64)
  for (int blk = 0; blk < NBLK; blk++) {
    int base = 4 + blk * 24;
    const float *ca_wk_b = PF(base + 0), *ca_wk_w = PF(base + 1);
    const float *ca_wo_b = PF(base + 2), *ca_wo_w = PF(base + 3);
    const float *ca_wq_b = PF(base + 4), *ca_wq_w = PF(base + 5);
    const float *ca_wv_b = PF(base + 6), *ca_wv_w = PF(base + 7);
    const float *fc1_b = PF(base + 8), *fc1_w = PF(base + 9);
    const float *fc2_b = PF(base + 10), *fc2_w = PF(base + 11);
    const float *fc3_b = PF(base + 12), *fc3_w = PF(base + 13);
    const float *mod_b = PF(base + 14), *mod_w = PF(base + 15);
    const float *sa_wk_b = PF(base + 16), *sa_wk_w = PF(base + 17);
    const float *sa_wo_b = PF(base + 18), *sa_wo_w = PF(base + 19);
    const float *sa_wq_b = PF(base + 20), *sa_wq_w = PF(base + 21);
    const float *sa_wv_b = PF(base + 22), *sa_wv_w = PF(base + 23);

    // modulation: silu(t_emb) @ mod  -> (B, 6144)
    k_small_linear<<<dim3(6 * HIDN / 64, BB), 64, 0, stream>>>(
        stb, mod_w, mod_b, modb, HIDN, 6 * HIDN, 0);

    // --- self attention ---
    k_ln<<<(int)TOK, 256, 0, stream>>>(x, modb + 0, modb + HIDN, 6 * HIDN, LL,
                                       HIDN, hb);
    gemm(hb, sa_wq_w, sa_wq_b, nullptr, qb, nullptr, nullptr, 0, (int)TOK,
         HIDN, HIDN, 0, scaleQ);
    gemm(hb, sa_wk_w, sa_wk_b, nullptr, kb, nullptr, nullptr, 0, (int)TOK,
         HIDN, HIDN, 0, 1.f);
    gemm(hb, sa_wv_w, sa_wv_b, nullptr, vb, nullptr, nullptr, 0, (int)TOK,
         HIDN, HIDN, 0, 1.f);
    k_attn<<<dim3(LL / 64, NHEADS, BB), 128, 0, stream>>>(qb, kb, vb, attnb,
                                                          nullptr, LL, LL);
    gemm(attnb, sa_wo_w, sa_wo_b, x, nullptr, x, modb + 2 * HIDN, 6 * HIDN,
         (int)TOK, HIDN, HIDN, 0, 1.f);

    // --- cross attention ---
    k_ln<<<(int)TOK, 256, 0, stream>>>(x, nullptr, nullptr, 0, LL, HIDN, hb);
    gemm(hb, ca_wq_w, ca_wq_b, nullptr, qb, nullptr, nullptr, 0, (int)TOK,
         HIDN, HIDN, 0, scaleQ);
    gemm(ctxB, ca_wk_w, ca_wk_b, nullptr, kb, nullptr, nullptr, 0, BB * LCC,
         HIDN, HIDN, 0, 1.f);
    gemm(ctxB, ca_wv_w, ca_wv_b, nullptr, vb, nullptr, nullptr, 0, BB * LCC,
         HIDN, HIDN, 0, 1.f);
    k_attn<<<dim3(LL / 64, NHEADS, BB), 128, 0, stream>>>(qb, kb, vb, attnb,
                                                          pad, LL, LCC);
    gemm(attnb, ca_wo_w, ca_wo_b, x, nullptr, x, nullptr, 0, (int)TOK, HIDN,
         HIDN, 0, 1.f);

    // --- MLP ---
    k_ln<<<(int)TOK, 256, 0, stream>>>(x, modb + 3 * HIDN, modb + 4 * HIDN,
                                       6 * HIDN, LL, HIDN, hb);
    gemm(hb, fc1_w, fc1_b, nullptr, m1, nullptr, nullptr, 0, (int)TOK, MLPH,
         HIDN, 1, 1.f);
    gemm(m1, fc2_w, fc2_b, nullptr, m2, nullptr, nullptr, 0, (int)TOK, MLPH,
         MLPH, 1, 1.f);
    gemm(m2, fc3_w, fc3_b, x, nullptr, x, modb + 5 * HIDN, 6 * HIDN, (int)TOK,
         HIDN, MLPH, 0, 1.f);
  }

  // --- final mod + LN + projection ---
  k_small_linear<<<dim3(2 * HIDN / 64, BB), 64, 0, stream>>>(
      stb, PF(103), PF(102), fmod, HIDN, 2 * HIDN, 0);
  k_ln<<<(int)TOK, 256, 0, stream>>>(x, fmod + 0, fmod + HIDN, 2 * HIDN, LL,
                                     HIDN, hb);
  gemm(hb, PF(105), PF(104), (float*)d_out, nullptr, nullptr, nullptr, 0,
       (int)TOK, LATD, HIDN, 0, 1.f);
}
